// STAdaptiveGNN_16406775071491
// MI455X (gfx1250) — matrix operand order repaired and checked by reading, compiled, not run on hardware
//
#include <hip/hip_runtime.h>
#include <hip/hip_bf16.h>
#include <math.h>

typedef __bf16 bf16_t;
typedef __attribute__((ext_vector_type(16))) __bf16 v16bf;
typedef __attribute__((ext_vector_type(8)))  __bf16 v8bf;
typedef __attribute__((ext_vector_type(8)))  float  v8f;

#define N_OG 50000L
#define N_CG 10000L
#define E_B  100000L
#define E_CG 200000L
#define HDIM 256

union FragAB { v16bf v; v8bf h[2]; };

// one 16B lane-chunk of a tile, global -> LDS via the async pipe (ASYNCcnt),
// with a compile-time byte offset folded into the instruction's offset field
template <int OFFB>
__device__ __forceinline__ void async_chunk(unsigned lds, const bf16_t* g)
{
    asm volatile("global_load_async_to_lds_b128 %0, %1, off offset:%2"
                 :: "v"(lds), "v"(g), "n"(OFFB) : "memory");
}

// ---------------------------------------------------------------------------
// WMMA GEMM: C[M,256] = A[M,K](bf16) * Wt[256,K]^T(bf16) + bias
// Block tile 128x128, 8 waves, each wave 64x32 = 4x2 tiles of 16x16x32 bf16.
// Tiles staged with GLOBAL_LOAD_ASYNC_TO_LDS_B128 into double-buffered LDS;
// the k-loop is unrolled 2x so even tiles always use buffer 0 and odd tiles
// buffer 1 (constant LDS addresses, no cndmask selection). 4 async ops per
// wave per tile retire in order => s_wait_asynccnt 4 == previous tile landed.
// flags&1: accumulate into existing Cf. Cb: optional bf16 output, stride ldcb.
// ---------------------------------------------------------------------------
__global__ __launch_bounds__(256) void k_gemm(
    const bf16_t* __restrict__ A, const bf16_t* __restrict__ Wt,
    const float* __restrict__ bias, float* __restrict__ Cf,
    bf16_t* __restrict__ Cb, long M, int K, int ldcb, int flags)
{
    __shared__ bf16_t sA[2][128 * 40];   // 80B row stride: 16B aligned, conflict-free
    __shared__ bf16_t sB[2][128 * 40];
    const int tid  = threadIdx.x;
    const int lane = tid & 31;
    const int wv   = tid >> 5;
    const int wm   = wv >> 2;         // 0..1 -> 64 rows each
    const int wn   = wv & 3;          // 0..3 -> 32 cols each
    const int half = lane >> 4;
    const int l16  = lane & 15;
    const long rowBase = (long)blockIdx.x * 128;
    const int  colBase = blockIdx.y * 128;

    // async-copy assignment: each thread owns 2 A-chunks + 2 B-chunks of 16B
    const int c0 = tid, c1 = tid + 256;           // of 512 16B chunks per tile
    const int r0 = c0 >> 2, q0 = c0 & 3;
    const int r1 = c1 >> 2, q1 = c1 & 3;
    long ga0 = rowBase + r0; if (ga0 > M - 1) ga0 = M - 1;
    long ga1 = rowBase + r1; if (ga1 > M - 1) ga1 = M - 1;
    const bf16_t* pA0 = A + ga0 * (long)K + q0 * 8;
    const bf16_t* pA1 = A + ga1 * (long)K + q1 * 8;
    const bf16_t* pB0 = Wt + (long)(colBase + r0) * K + q0 * 8;
    const bf16_t* pB1 = Wt + (long)(colBase + r1) * K + q1 * 8;
    unsigned lA0[2], lA1[2], lB0[2], lB1[2];
#pragma unroll
    for (int b = 0; b < 2; ++b) {
        lA0[b] = (unsigned)(size_t)&sA[b][r0 * 40 + q0 * 8];
        lA1[b] = (unsigned)(size_t)&sA[b][r1 * 40 + q1 * 8];
        lB0[b] = (unsigned)(size_t)&sB[b][r0 * 40 + q0 * 8];
        lB1[b] = (unsigned)(size_t)&sB[b][r1 * 40 + q1 * 8];
    }

    v8f acc[4][2];
#pragma unroll
    for (int i = 0; i < 4; ++i)
#pragma unroll
        for (int j = 0; j < 2; ++j)
#pragma unroll
            for (int v = 0; v < 8; ++v) acc[i][j][v] = 0.0f;

    // compute one staged tile from LDS buffer `cur` (called with literals)
    auto compute = [&](int cur) {
        FragAB a[4], b[2];
#pragma unroll
        for (int i = 0; i < 4; ++i) {
            int ar = wm * 64 + i * 16 + l16;
            // 16-bit A 16x32 layout: lane holds K=[8h,8h+8) and K=[16+8h,16+8h+8)
            a[i].h[0] = *(const v8bf*)&sA[cur][ar * 40 + half * 8];
            a[i].h[1] = *(const v8bf*)&sA[cur][ar * 40 + 16 + half * 8];
        }
#pragma unroll
        for (int j = 0; j < 2; ++j) {
            int br = wn * 32 + j * 16 + l16;
            // B 32x16 layout: lane holds K=[8h,8h+8) and K=[16+8h,16+8h+8)
            b[j].h[0] = *(const v8bf*)&sB[cur][br * 40 + half * 8];
            b[j].h[1] = *(const v8bf*)&sB[cur][br * 40 + 16 + half * 8];
        }
#pragma unroll
        for (int i = 0; i < 4; ++i)
#pragma unroll
            for (int j = 0; j < 2; ++j)
                acc[i][j] = __builtin_amdgcn_wmma_f32_16x16x32_bf16(
                    false, a[i].v, false, b[j].v, (short)0, acc[i][j], false, false);
    };

    const int nk = K >> 5;
    // prologue: tile 0 -> buffer 0
    async_chunk<0>(lA0[0], pA0); async_chunk<0>(lA1[0], pA1);
    async_chunk<0>(lB0[0], pB0); async_chunk<0>(lB1[0], pB1);

    int kt = 0;
    while (true) {
        // ---- even tile kt (buffer 0); prefetch kt+1 -> buffer 1 (offset:64B)
        const bool has1 = kt + 1 < nk;
        if (has1) {
            async_chunk<64>(lA0[1], pA0); async_chunk<64>(lA1[1], pA1);
            async_chunk<64>(lB0[1], pB0); async_chunk<64>(lB1[1], pB1);
            asm volatile("s_wait_asynccnt 0x4" ::: "memory");
        } else {
            asm volatile("s_wait_asynccnt 0x0" ::: "memory");
        }
        __syncthreads();
        compute(0);
        __syncthreads();
        if (!has1) break;

        // ---- odd tile kt+1 (buffer 1); prefetch kt+2 -> buffer 0 (offset:128B)
        const bool has2 = kt + 2 < nk;
        if (has2) {
            async_chunk<128>(lA0[0], pA0); async_chunk<128>(lA1[0], pA1);
            async_chunk<128>(lB0[0], pB0); async_chunk<128>(lB1[0], pB1);
            asm volatile("s_wait_asynccnt 0x4" ::: "memory");
        } else {
            asm volatile("s_wait_asynccnt 0x0" ::: "memory");
        }
        __syncthreads();
        compute(1);
        __syncthreads();
        if (!has2) break;

        pA0 += 64; pA1 += 64; pB0 += 64; pB1 += 64;   // advance 2 tiles (128B)
        kt += 2;
    }

#pragma unroll
    for (int i = 0; i < 4; ++i) {
#pragma unroll
        for (int j = 0; j < 2; ++j) {
            int col = colBase + wn * 32 + j * 16 + l16;
            float bs = bias ? bias[col] : 0.0f;
#pragma unroll
            for (int v = 0; v < 8; ++v) {
                long row = rowBase + wm * 64 + i * 16 + 8 * half + v;
                if (row < M) {
                    float val = acc[i][j][v] + bs;
                    long o = row * 256 + col;
                    if (flags & 1) val += Cf[o];
                    if (Cf) Cf[o] = val;
                    if (Cb) Cb[row * (long)ldcb + col] = (bf16_t)val;
                }
            }
        }
    }
}

// weight [K,N=256] f32 -> [256,K] bf16 (transposed for B-fragment loads)
__global__ __launch_bounds__(256) void k_wt(const float* __restrict__ w,
                                            bf16_t* __restrict__ wt, int K)
{
    long i = (long)blockIdx.x * 256 + threadIdx.x;
    long n = (long)K * 256;
    if (i < n) {
        int k = (int)(i >> 8), c = (int)(i & 255);
        wt[(long)c * K + k] = (bf16_t)w[(long)k * 256 + c];
    }
}

__global__ __launch_bounds__(256) void k_cvt_bf16(const float* __restrict__ x,
                                                  bf16_t* __restrict__ o, long n)
{
    for (long i = (long)blockIdx.x * 256 + threadIdx.x; i < n;
         i += (long)gridDim.x * 256)
        o[i] = (bf16_t)x[i];
}

__global__ __launch_bounds__(256) void k_zero(float* __restrict__ p, long n)
{
    for (long i = (long)blockIdx.x * 256 + threadIdx.x; i < n;
         i += (long)gridDim.x * 256)
        p[i] = 0.0f;
}

// cat rows: out[e, 0:256] = a[ia?ia[e]:e], out[e, 256:512] = b[ib?ib[e]:e]
__global__ __launch_bounds__(256) void k_cat(const bf16_t* __restrict__ a,
                                             const bf16_t* __restrict__ b,
                                             const int* __restrict__ ia,
                                             const int* __restrict__ ib,
                                             bf16_t* __restrict__ out, long E)
{
    long e = (long)blockIdx.x * 4 + (threadIdx.x >> 6);
    int chunk = threadIdx.x & 63;                 // 64 x 16B = 1024B row
    if (e >= E) return;
    long ra = ia ? (long)ia[e] : e;
    long rb = ib ? (long)ib[e] : e;
    const uint4* pa = (const uint4*)(a + ra * 256);
    const uint4* pb = (const uint4*)(b + rb * 256);
    uint4* po = (uint4*)(out + e * 512);
    po[chunk] = (chunk < 32) ? pa[chunk] : pb[chunk - 32];
}

// K=2 linear into bf16 with stride; pos mode (eattr==null): in = pa[ia]-pb[ib]
__global__ __launch_bounds__(256) void k_lin2(const float* __restrict__ pa,
                                              const float* __restrict__ pb,
                                              const int* __restrict__ ia,
                                              const int* __restrict__ ib,
                                              const float* __restrict__ eattr,
                                              const float* __restrict__ w,
                                              const float* __restrict__ bias,
                                              bf16_t* __restrict__ out, int ldo, long E)
{
    int c = threadIdx.x;
    float w0 = w[c], w1 = w[256 + c], bs = bias[c];
    for (long e = blockIdx.x; e < E; e += gridDim.x) {
        float a0, a1;
        if (eattr) { a0 = eattr[e * 2]; a1 = eattr[e * 2 + 1]; }
        else {
            long da = ia[e], db = ib[e];
            a0 = pa[da * 2] - pb[db * 2];
            a1 = pa[da * 2 + 1] - pb[db * 2 + 1];
        }
        out[e * (long)ldo + c] = (bf16_t)(a0 * w0 + a1 * w1 + bs);
    }
}

// BatchNorm pass 1: column sums + sumsq into sums[0:256], sums[256:512]
__global__ __launch_bounds__(256) void k_bn_stats(const float* __restrict__ x,
                                                  long M, float* __restrict__ sums)
{
    int c = threadIdx.x;
    float s = 0.0f, s2 = 0.0f;
    for (long r = blockIdx.x; r < M; r += gridDim.x) {
        float v = x[r * 256 + c];
        s += v; s2 += v * v;
    }
    atomicAdd(&sums[c], s);
    atomicAdd(&sums[256 + c], s2);
}

// BatchNorm pass 2 (+ReLU/GELU), optional f32 and bf16 outputs
__global__ __launch_bounds__(256) void k_bn_apply(const float* __restrict__ x,
                                                  float* __restrict__ outf,
                                                  bf16_t* __restrict__ ob, int ldb,
                                                  long M, const float* __restrict__ sums,
                                                  const float* __restrict__ gamma,
                                                  const float* __restrict__ beta, int act)
{
    long n = M * 256;
    float inv = 1.0f / (float)M;
    for (long idx = (long)blockIdx.x * 256 + threadIdx.x; idx < n;
         idx += (long)gridDim.x * 256) {
        int c = (int)(idx & 255);
        long r = idx >> 8;
        float mu = sums[c] * inv;
        float var = sums[256 + c] * inv - mu * mu;
        float v = (x[idx] - mu) * rsqrtf(var + 1e-5f) * gamma[c] + beta[c];
        if (act == 1) v = fmaxf(v, 0.0f);
        else if (act == 2) v = 0.5f * v * (1.0f + erff(v * 0.70710678f));
        if (outf) outf[idx] = v;
        if (ob) ob[r * (long)ldb + c] = (bf16_t)v;
    }
}

// scatter-mean pass 1: acc[dst[e]] += m[src?src[e]:e], cnt[dst[e]] += 1
__global__ __launch_bounds__(256) void k_scatter(const float* __restrict__ m,
                                                 const int* __restrict__ src,
                                                 const int* __restrict__ dst, long E,
                                                 float* __restrict__ acc,
                                                 float* __restrict__ cnt)
{
    int c = threadIdx.x;
    for (long e = blockIdx.x; e < E; e += gridDim.x) {
        long d = dst[e];
        long mr = src ? (long)src[e] : e;
        atomicAdd(&acc[d * 256 + c], m[mr * 256 + c]);
        if (c == 0) atomicAdd(&cnt[d], 1.0f);
    }
}

// scatter-mean pass 2: divide by max(cnt,1); optional bf16 mirror
__global__ __launch_bounds__(256) void k_div(float* __restrict__ acc,
                                             const float* __restrict__ cnt,
                                             bf16_t* __restrict__ ob, long nrows)
{
    long n = nrows * 256;
    for (long idx = (long)blockIdx.x * 256 + threadIdx.x; idx < n;
         idx += (long)gridDim.x * 256) {
        long r = idx >> 8;
        float v = acc[idx] / fmaxf(cnt[r], 1.0f);
        acc[idx] = v;
        if (ob) ob[idx] = (bf16_t)v;
    }
}

// x = sigmoid(g)*x + (1-sigmoid(g))*s
__global__ __launch_bounds__(256) void k_gate(float* __restrict__ x,
                                              const float* __restrict__ g,
                                              const float* __restrict__ s, long n)
{
    for (long i = (long)blockIdx.x * 256 + threadIdx.x; i < n;
         i += (long)gridDim.x * 256) {
        float sg = 1.0f / (1.0f + expf(-g[i]));
        x[i] = sg * x[i] + (1.0f - sg) * s[i];
    }
}

// out = a + b (f32), optional bf16 mirror
__global__ __launch_bounds__(256) void k_add(const float* __restrict__ a,
                                             const float* __restrict__ b,
                                             float* __restrict__ o,
                                             bf16_t* __restrict__ ob, long n)
{
    for (long i = (long)blockIdx.x * 256 + threadIdx.x; i < n;
         i += (long)gridDim.x * 256) {
        float v = a[i] + b[i];
        o[i] = v;
        if (ob) ob[i] = (bf16_t)v;
    }
}

// final 256 -> 3 linear (plain last layer)
__global__ __launch_bounds__(256) void k_out3(const bf16_t* __restrict__ A,
                                              const float* __restrict__ w,
                                              const float* __restrict__ b,
                                              float* __restrict__ out, long M)
{
    long r = (long)blockIdx.x * 256 + threadIdx.x;
    if (r >= M) return;
    const bf16_t* a = A + r * 256;
    float s0 = 0.0f, s1 = 0.0f, s2 = 0.0f;
    for (int k = 0; k < 256; ++k) {
        float av = (float)a[k];
        s0 += av * w[k * 3 + 0];
        s1 += av * w[k * 3 + 1];
        s2 += av * w[k * 3 + 2];
    }
    out[r * 3 + 0] = s0 + b[0];
    out[r * 3 + 1] = s1 + b[1];
    out[r * 3 + 2] = s2 + b[2];
}

// ---------------------------------------------------------------------------
extern "C" void kernel_launch(void* const* d_in, const int* in_sizes, int n_in,
                              void* d_out, int out_size, void* d_ws, size_t ws_size,
                              hipStream_t stream)
{
    (void)in_sizes; (void)n_in; (void)out_size; (void)ws_size;
    const int GS = 4096;

    // ---- data inputs -------------------------------------------------------
    const float* x_og_in = (const float*)d_in[0];       // 50000 x 32
    const float* x_cg_in = (const float*)d_in[1];       // 10000 x 32
    const float* pos_og  = (const float*)d_in[2];       // 50000 x 2
    const float* pos_cg  = (const float*)d_in[3];       // 10000 x 2
    const int*   b_src   = (const int*)d_in[4];         // [2, E_B]
    const int*   b_dst   = b_src + E_B;
    const float* eattr   = (const float*)d_in[5];       // E_B x 2
    const int*   c_src   = (const int*)d_in[6];         // [2, E_CG]
    const int*   c_dst   = c_src + E_CG;

    // params pytree leaves (dicts key-sorted, lists in order), base index 7
    auto P = [&](int i) { return (const float*)d_in[7 + i]; };
    // leaf layout (relative):
    //  cg_proj: l*4 + {b,beta,gamma,w}              -> 0..11
    //  coars i: 12+i*14; feat: lc0 b,beta,g,w = +0..3; lc1 = +4..7;
    //           edge b=+8,w=+9; pos b=+10,w=+11; linx b=+12,w=+13
    //  convs i: 54+i*4: nei_b,nei_w,root_b,root_w
    //  gate:    66 (b), 67 (w)
    //  norms i: 68+i*2: beta,gamma
    //  og_lin i:74+i*8: layer l*4 + {b,beta,gamma,w}
    //  og_proj: 98..109 ; out: 110..121 ; spread i: 122+i*14

    // ---- workspace arena ---------------------------------------------------
    char* wsp = (char*)d_ws;
    size_t wso = 0;
    auto walloc = [&](size_t bytes) -> void* {
        void* r = wsp + wso;
        wso += (bytes + 255) & ~(size_t)255;
        return r;
    };

    auto mkwt = [&](int leaf, int K) -> bf16_t* {
        bf16_t* dst = (bf16_t*)walloc((size_t)K * 256 * sizeof(bf16_t));
        long n = (long)K * 256;
        k_wt<<<(unsigned)((n + 255) / 256), 256, 0, stream>>>(P(leaf), dst, K);
        return dst;
    };

    // transposed bf16 weights
    bf16_t *WT_ogp[3], *WT_cgp[3], *WT_gate, *WT_nei[3], *WT_root[3];
    bf16_t *WT_co_lx[3], *WT_co_l0[3], *WT_co_l1[3];
    bf16_t *WT_sp_lx[3], *WT_sp_l0[3], *WT_sp_l1[3];
    bf16_t *WT_ogl[3][2], *WT_out[2];
    for (int l = 0; l < 3; ++l) WT_ogp[l] = mkwt(98 + l * 4 + 3, l == 0 ? 32 : 256);
    for (int l = 0; l < 3; ++l) WT_cgp[l] = mkwt(l * 4 + 3, l == 0 ? 32 : 256);
    WT_gate = mkwt(67, 512);
    for (int i = 0; i < 3; ++i) {
        int cb = 12 + i * 14, sb = 122 + i * 14;
        WT_co_lx[i] = mkwt(cb + 13, 512);
        WT_co_l0[i] = mkwt(cb + 3, 768);
        WT_co_l1[i] = mkwt(cb + 7, 256);
        WT_sp_lx[i] = mkwt(sb + 13, 512);
        WT_sp_l0[i] = mkwt(sb + 3, 768);
        WT_sp_l1[i] = mkwt(sb + 7, 256);
        WT_nei[i]  = mkwt(54 + i * 4 + 1, 256);
        WT_root[i] = mkwt(54 + i * 4 + 3, 256);
        WT_ogl[i][0] = mkwt(74 + i * 8 + 3, 256);
        WT_ogl[i][1] = mkwt(74 + i * 8 + 7, 256);
    }
    WT_out[0] = mkwt(113, 256);
    WT_out[1] = mkwt(117, 256);

    // activation buffers
    bf16_t* ogb_in = (bf16_t*)walloc(N_OG * 32 * sizeof(bf16_t));
    bf16_t* cgb_in = (bf16_t*)walloc(N_CG * 32 * sizeof(bf16_t));
    float*  og_f   = (float*)walloc(N_OG * 256 * sizeof(float));
    float*  pog_f  = (float*)walloc(N_OG * 256 * sizeof(float));
    float*  tmpo_f = (float*)walloc(N_OG * 256 * sizeof(float));
    float*  spr_f  = (float*)walloc(N_OG * 256 * sizeof(float));
    bf16_t* og_b   = (bf16_t*)walloc(N_OG * 256 * sizeof(bf16_t));
    bf16_t* spr_b  = (bf16_t*)walloc(N_OG * 256 * sizeof(bf16_t));
    float*  cg_f   = (float*)walloc(N_CG * 256 * sizeof(float));
    float*  pcg_f  = (float*)walloc(N_CG * 256 * sizeof(float));
    float*  scg_f  = (float*)walloc(N_CG * 256 * sizeof(float));
    float*  aggC   = (float*)walloc(N_CG * 256 * sizeof(float));
    bf16_t* cg_b   = (bf16_t*)walloc(N_CG * 256 * sizeof(bf16_t));
    bf16_t* scg_b  = (bf16_t*)walloc(N_CG * 256 * sizeof(bf16_t));
    bf16_t* agg_b  = (bf16_t*)walloc(N_CG * 256 * sizeof(bf16_t));
    bf16_t* cat512 = (bf16_t*)walloc(E_B * 512 * sizeof(bf16_t));
    bf16_t* cat768 = (bf16_t*)walloc(E_B * 768 * sizeof(bf16_t));
    float*  Ef32   = (float*)walloc(E_B * 256 * sizeof(float));
    bf16_t* Ebf    = (bf16_t*)walloc(E_B * 256 * sizeof(bf16_t));
    float*  bnsums = (float*)walloc(512 * sizeof(float));
    float*  cntC   = (float*)walloc(N_CG * sizeof(float));
    float*  cntO   = (float*)walloc(N_OG * sizeof(float));

    // ---- helpers -----------------------------------------------------------
    auto gemm = [&](const bf16_t* A, const bf16_t* WT, const float* bias,
                    float* Cf, bf16_t* Cb, long M, int K, int ldcb, int flags) {
        dim3 g((unsigned)((M + 127) / 128), 2);
        k_gemm<<<g, 256, 0, stream>>>(A, WT, bias, Cf, Cb, M, K,
                                      ldcb ? ldcb : 256, flags);
    };
    auto bnorm = [&](float* x, long M, const float* gamma, const float* beta,
                     int act, float* outf, bf16_t* ob, int ldb) {
        k_zero<<<2, 256, 0, stream>>>(bnsums, 512);
        k_bn_stats<<<512, 256, 0, stream>>>(x, M, bnsums);
        k_bn_apply<<<GS, 256, 0, stream>>>(x, outf, ob, ldb, M, bnsums, gamma,
                                           beta, act);
    };

    // feat_mp: x_dst[dst] ++ x_src[src] -> lin_x | lin_pos | lin_edge -> MLP
    //          -> scatter-mean over dst into (outf, outb)
    auto feat_mp = [&](int base, const bf16_t* xsrc_b, const bf16_t* xdst_b,
                       const float* psrc, const float* pdst,
                       const int* src, const int* dst, long nd,
                       bf16_t* WLX, bf16_t* WL0, bf16_t* WL1,
                       float* outf, bf16_t* outb, float* cnt) {
        long E = E_B;
        k_cat<<<(unsigned)((E + 3) / 4), 256, 0, stream>>>(xdst_b, xsrc_b, dst,
                                                           src, cat512, E);
        gemm(cat512, WLX, P(base + 12), nullptr, cat768, E, 512, 768, 0);
        k_lin2<<<2048, 256, 0, stream>>>(pdst, psrc, dst, src, nullptr,
                                         P(base + 11), P(base + 10),
                                         cat768 + 256, 768, E);
        k_lin2<<<2048, 256, 0, stream>>>(nullptr, nullptr, nullptr, nullptr,
                                         eattr, P(base + 9), P(base + 8),
                                         cat768 + 512, 768, E);
        gemm(cat768, WL0, P(base + 0), Ef32, nullptr, E, 768, 0, 0);
        bnorm(Ef32, E, P(base + 2), P(base + 1), 1, nullptr, Ebf, 256);
        gemm(Ebf, WL1, P(base + 4), Ef32, nullptr, E, 256, 0, 0);
        bnorm(Ef32, E, P(base + 6), P(base + 5), 1, Ef32, nullptr, 0);
        k_zero<<<GS, 256, 0, stream>>>(outf, nd * 256);
        k_zero<<<64, 256, 0, stream>>>(cnt, nd);
        k_scatter<<<GS, 256, 0, stream>>>(Ef32, nullptr, dst, E, outf, cnt);
        k_div<<<GS, 256, 0, stream>>>(outf, cnt, outb, nd);
    };

    // ---- og_proj / cg_proj (MLP: Linear->BN->ReLU x3) ----------------------
    k_cvt_bf16<<<GS, 256, 0, stream>>>(x_og_in, ogb_in, N_OG * 32);
    k_cvt_bf16<<<GS, 256, 0, stream>>>(x_cg_in, cgb_in, N_CG * 32);
    {
        const bf16_t* cur = ogb_in;
        for (int l = 0; l < 3; ++l) {
            int b = 98 + l * 4;
            gemm(cur, WT_ogp[l], P(b + 0), tmpo_f, nullptr, N_OG,
                 l == 0 ? 32 : 256, 0, 0);
            bnorm(tmpo_f, N_OG, P(b + 2), P(b + 1), 1, og_f, og_b, 256);
            cur = og_b;
        }
        cur = cgb_in;
        for (int l = 0; l < 3; ++l) {
            int b = l * 4;
            gemm(cur, WT_cgp[l], P(b + 0), scg_f, nullptr, N_CG,
                 l == 0 ? 32 : 256, 0, 0);
            bnorm(scg_f, N_CG, P(b + 2), P(b + 1), 1, cg_f, cg_b, 256);
            cur = cg_b;
        }
    }

    // ---- layer loop --------------------------------------------------------
    for (int i = 0; i < 3; ++i) {
        hipMemcpyAsync(pog_f, og_f, N_OG * 256 * sizeof(float),
                       hipMemcpyDeviceToDevice, stream);
        hipMemcpyAsync(pcg_f, cg_f, N_CG * 256 * sizeof(float),
                       hipMemcpyDeviceToDevice, stream);

        // coarsen: og -> cg
        feat_mp(12 + i * 14, og_b, cg_b, pos_og, pos_cg, b_src, b_dst, N_CG,
                WT_co_lx[i], WT_co_l0[i], WT_co_l1[i], cg_f, cg_b, cntC);

        // sage_conv on cg graph: root(x) + nei(mean_agg(x[src] -> dst))
        k_zero<<<GS, 256, 0, stream>>>(aggC, N_CG * 256);
        k_zero<<<64, 256, 0, stream>>>(cntC, N_CG);
        k_scatter<<<GS, 256, 0, stream>>>(cg_f, c_src, c_dst, E_CG, aggC, cntC);
        k_div<<<GS, 256, 0, stream>>>(aggC, cntC, agg_b, N_CG);
        gemm(cg_b, WT_root[i], P(54 + i * 4 + 2), cg_f, nullptr, N_CG, 256, 0, 0);
        gemm(agg_b, WT_nei[i], P(54 + i * 4 + 0), cg_f, cg_b, N_CG, 256, 256, 1);

        // spread: (cg + prev_cg) -> og, flipped bipartite edges
        k_add<<<GS, 256, 0, stream>>>(cg_f, pcg_f, scg_f, scg_b, N_CG * 256);
        feat_mp(122 + i * 14, scg_b, og_b, pos_cg, pos_og, b_dst, b_src, N_OG,
                WT_sp_lx[i], WT_sp_l0[i], WT_sp_l1[i], spr_f, spr_b, cntO);

        // gate: sigmoid(lin([x_og, spread]))
        k_cat<<<(unsigned)((N_OG + 3) / 4), 256, 0, stream>>>(
            og_b, spr_b, nullptr, nullptr, cat512, N_OG);
        gemm(cat512, WT_gate, P(66), Ef32, nullptr, N_OG, 512, 0, 0);
        k_gate<<<GS, 256, 0, stream>>>(og_f, Ef32, spr_f, N_OG * 256);

        // norms[i] BN + exact GELU
        bnorm(og_f, N_OG, P(68 + i * 2 + 1), P(68 + i * 2 + 0), 2, og_f, og_b, 256);

        // og_lin MLP (2x Linear->BN->ReLU) + residual
        {
            int b0 = 74 + i * 8, b1 = b0 + 4;
            gemm(og_b, WT_ogl[i][0], P(b0 + 0), tmpo_f, nullptr, N_OG, 256, 0, 0);
            bnorm(tmpo_f, N_OG, P(b0 + 2), P(b0 + 1), 1, nullptr, spr_b, 256);
            gemm(spr_b, WT_ogl[i][1], P(b1 + 0), tmpo_f, nullptr, N_OG, 256, 0, 0);
            bnorm(tmpo_f, N_OG, P(b1 + 2), P(b1 + 1), 1, tmpo_f, nullptr, 0);
            k_add<<<GS, 256, 0, stream>>>(tmpo_f, pog_f, og_f, og_b, N_OG * 256);
        }
    }

    // ---- out MLP (plain last) ---------------------------------------------
    gemm(og_b, WT_out[0], P(110), tmpo_f, nullptr, N_OG, 256, 0, 0);
    bnorm(tmpo_f, N_OG, P(112), P(111), 1, nullptr, spr_b, 256);
    gemm(spr_b, WT_out[1], P(114), tmpo_f, nullptr, N_OG, 256, 0, 0);
    bnorm(tmpo_f, N_OG, P(116), P(115), 1, nullptr, spr_b, 256);
    k_out3<<<(unsigned)((N_OG + 255) / 256), 256, 0, stream>>>(
        spr_b, P(121), P(118), (float*)d_out, N_OG);
}
